// FlashGPT2Attention_83391085019898
// MI455X (gfx1250) — matrix operand order repaired and checked by reading, not compiled, not run on hardware
//
#include <hip/hip_runtime.h>
#include <hip/hip_bf16.h>
#include <math.h>

// ---------------------------------------------------------------------------
// GPT-2 attention block for MI455X (gfx1250), BF16 WMMA + fp32 accumulate.
// B=4, S=1024, H=16, D=128, E=2048, 3E=6144, TOTAL=4096.
// Compute-bound workload (~170 GFLOP vs ~150 MB traffic) -> everything runs
// through v_wmma_f32_16x16x32_bf16 with fp32 accumulators.
// ---------------------------------------------------------------------------

typedef __attribute__((ext_vector_type(16))) __bf16 v16bf;
typedef __attribute__((ext_vector_type(8)))  __bf16 v8bf;
typedef __attribute__((ext_vector_type(8)))  float  v8f;
typedef __attribute__((ext_vector_type(4)))  float  v4f;

static __device__ inline v16bf cat16(v8bf lo, v8bf hi) {
  return __builtin_shufflevector(lo, hi, 0,1,2,3,4,5,6,7,8,9,10,11,12,13,14,15);
}

static __device__ inline v8f wmma_bf16(v16bf a, v16bf b, v8f c) {
  // D = A(16x32 bf16) * B(32x16 bf16) + C(16x16 f32)
  return __builtin_amdgcn_wmma_f32_16x16x32_bf16(false, a, false, b, (short)0, c,
                                                 false, false);
}

static __device__ inline void wave_lds_fence() {
  // LDS ops from one wave execute in order (ISA Ch.11); this only stops the
  // compiler from reordering the cross-lane LDS read before the write.
  __builtin_amdgcn_wave_barrier();
  asm volatile("" ::: "memory");
}

#define BM 128
#define BN 128
#define BK 32

// ---------------------------------------------------------------------------
// C[M,N] = A[M,K] (f32,row-major) x B[K,N] (f32,row-major) + bias[N], f32 out.
// 256 threads = 8 waves (4x2 wave grid); wave tile = 32x64 = 2x4 WMMA tiles.
// Double-buffered LDS: one barrier per K-step; next tile's global loads are
// issued before the barrier so they overlap the 8 WMMAs of the current tile.
// ---------------------------------------------------------------------------
__global__ __launch_bounds__(256)
void gemm_bias_bf16wmma(const float* __restrict__ A, const float* __restrict__ Bm,
                        const float* __restrict__ bias, float* __restrict__ C,
                        int M, int N, int K) {
  __shared__ __align__(32) __bf16 As[2][BM * BK];  // 2 x 8 KB
  __shared__ __align__(32) __bf16 Bs[2][BK * BN];  // 2 x 8 KB

  const int tid  = threadIdx.x;
  const int lane = tid & 31;
  const int wave = tid >> 5;
  const int wm   = wave >> 1;            // 0..3 : 32 rows each
  const int wn   = wave & 1;             // 0..1 : 64 cols each
  const int row0 = blockIdx.y * BM;
  const int col0 = blockIdx.x * BN;
  const int m    = lane & 15;
  const int half = lane >> 4;
  const int kb   = 8 * half;             // per-half K base (A-fragment layout)

  v8f zero = {};
  v8f acc[2][4];
#pragma unroll
  for (int mi = 0; mi < 2; ++mi)
#pragma unroll
    for (int ni = 0; ni < 4; ++ni) acc[mi][ni] = zero;

  // ---- stage tile 0 into buffer 0 (f32 -> bf16)
#pragma unroll
  for (int i = 0; i < 4; ++i) {
    int g = tid + i * 256;
    {
      int r = g >> 3, c = (g & 7) * 4;   // 128 rows x 8 col-quads
      v4f v = *(const v4f*)(A + (size_t)(row0 + r) * K + c);
      __bf16* d = &As[0][r * BK + c];
      d[0] = (__bf16)v.x; d[1] = (__bf16)v.y; d[2] = (__bf16)v.z; d[3] = (__bf16)v.w;
    }
    {
      int r = g >> 5, c = (g & 31) * 4;  // 32 rows x 32 col-quads
      v4f v = *(const v4f*)(Bm + (size_t)r * N + col0 + c);
      __bf16* d = &Bs[0][r * BN + c];
      d[0] = (__bf16)v.x; d[1] = (__bf16)v.y; d[2] = (__bf16)v.z; d[3] = (__bf16)v.w;
    }
  }

  int buf = 0;
  for (int kt = 0; kt < K; kt += BK) {
    const bool more = (kt + BK) < K;

    // ---- issue next-tile global loads first (latency hidden behind WMMAs)
    v4f an[4], bn[4];
    if (more) {
#pragma unroll
      for (int i = 0; i < 4; ++i) {
        int g = tid + i * 256;
        an[i] = *(const v4f*)(A + (size_t)(row0 + (g >> 3)) * K + kt + BK + (g & 7) * 4);
        bn[i] = *(const v4f*)(Bm + (size_t)(kt + BK + (g >> 5)) * N + col0 + (g & 31) * 4);
      }
      if (kt + 2 * BK < K) {  // warm L2 two tiles ahead -> global_prefetch_b8
        __builtin_prefetch(A + (size_t)(row0 + (tid >> 1)) * K + kt + 2 * BK, 0, 1);
        __builtin_prefetch(Bm + (size_t)(kt + 2 * BK + (tid & 31)) * N + col0, 0, 1);
      }
    }

    __syncthreads();  // staging of As[buf]/Bs[buf] complete, prior reads done

    // ---- all fragments up front, then 8 back-to-back WMMAs
    const __bf16* Ac = As[buf];
    const __bf16* Bc = Bs[buf];
    v16bf afr[2], bfr[4];
#pragma unroll
    for (int mi = 0; mi < 2; ++mi) {
      int row = (wm * 2 + mi) * 16 + m;
      afr[mi] = cat16(*(const v8bf*)&Ac[row * BK + kb],
                      *(const v8bf*)&Ac[row * BK + kb + 16]);
    }
#pragma unroll
    for (int ni = 0; ni < 4; ++ni) {
      int coln = (wn * 4 + ni) * 16;
#pragma unroll
      for (int s = 0; s < 16; ++s)
        bfr[ni][s] = Bc[(kb + (s & 7) + ((s >> 3) << 4)) * BN + coln + m];
    }
#pragma unroll
    for (int ni = 0; ni < 4; ++ni)
#pragma unroll
      for (int mi = 0; mi < 2; ++mi)
        acc[mi][ni] = wmma_bf16(afr[mi], bfr[ni], acc[mi][ni]);

    // ---- convert + store next tile into the other buffer
    if (more) {
      __bf16* An = As[buf ^ 1];
      __bf16* Bn = Bs[buf ^ 1];
#pragma unroll
      for (int i = 0; i < 4; ++i) {
        int g = tid + i * 256;
        {
          int r = g >> 3, c = (g & 7) * 4;
          __bf16* d = &An[r * BK + c];
          d[0] = (__bf16)an[i].x; d[1] = (__bf16)an[i].y;
          d[2] = (__bf16)an[i].z; d[3] = (__bf16)an[i].w;
        }
        {
          int r = g >> 5, c = (g & 31) * 4;
          __bf16* d = &Bn[r * BN + c];
          d[0] = (__bf16)bn[i].x; d[1] = (__bf16)bn[i].y;
          d[2] = (__bf16)bn[i].z; d[3] = (__bf16)bn[i].w;
        }
      }
    }
    buf ^= 1;
  }

  // ---- epilogue: C/D layout -> lane(row = r + 8*half, col = lane&15)
#pragma unroll
  for (int mi = 0; mi < 2; ++mi)
#pragma unroll
    for (int ni = 0; ni < 4; ++ni)
#pragma unroll
      for (int r = 0; r < 8; ++r) {
        int row = row0 + (wm * 2 + mi) * 16 + r + 8 * half;
        int col = col0 + (wn * 4 + ni) * 16 + (lane & 15);
        C[(size_t)row * N + col] = acc[mi][ni][r] + bias[col];
      }
}

// ---------------------------------------------------------------------------
// Flash attention, causal. Grid: (S/128, B*H). 8 waves; wave owns 16 q rows.
// Q fragments in registers; K^T and V tiles (32 keys) staged in LDS as bf16;
// fp32 online softmax on the WMMA accumulator layout.
// ---------------------------------------------------------------------------
#define SQ 1024
#define DD 128
#define EE 2048
#define E3 6144

__global__ __launch_bounds__(256)
void flash_attn_bf16wmma(const float* __restrict__ qkv, float* __restrict__ ctx) {
  __shared__ __align__(32) __bf16 Kt[DD * 32];     // [d][kk]   8 KB
  __shared__ __align__(32) __bf16 Vs[32 * DD];     // [kk][d]   8 KB
  __shared__ __align__(32) __bf16 Ps[8][16 * 32];  // per-wave  8 KB

  const int tid  = threadIdx.x;
  const int lane = tid & 31;
  const int wave = tid >> 5;
  const int qblk = blockIdx.x;          // 0..7
  const int bh   = blockIdx.y;
  const int b    = bh >> 4;
  const int h    = bh & 15;
  const int m    = lane & 15;
  const int half = lane >> 4;
  const int kb   = 8 * half;
  const int q0   = qblk * 128 + wave * 16;

  // ---- resident Q fragments: 4 K-slices of 32 over D=128
  v16bf qfr[4];
#pragma unroll
  for (int j = 0; j < 4; ++j) {
    const float* qp = qkv + (size_t)(b * SQ + q0 + m) * E3 + h * DD + j * 32;
    v16bf f;
#pragma unroll
    for (int e = 0; e < 8; ++e) f[e] = (__bf16)qp[kb + e];
#pragma unroll
    for (int e = 0; e < 8; ++e) f[8 + e] = (__bf16)qp[kb + 16 + e];
    qfr[j] = f;
  }

  v8f zero = {};
  v8f oacc[8];
#pragma unroll
  for (int dt = 0; dt < 8; ++dt) oacc[dt] = zero;
  float mrun[8], lrun[8];
#pragma unroll
  for (int r = 0; r < 8; ++r) { mrun[r] = -3.0e38f; lrun[r] = 0.f; }

  const float scale = 0.08838834764831845f;  // 1/sqrt(128)
  const int nT = (qblk + 1) * 4;             // 32-key tiles up to block end

  for (int t = 0; t < nT; ++t) {
    const int k0 = t * 32;
    // ---- warm L2 for the next tile's K and V rows
    if (t + 1 < nT) {
      const float* nk =
          qkv + (size_t)(b * SQ + k0 + 32 + (tid & 31)) * E3 + EE + h * DD;
      __builtin_prefetch(nk, 0, 1);
      __builtin_prefetch(nk + EE, 0, 1);
    }
    // ---- stage K (transposed -> Kt[d][kk]) and V (Vs[kk][d]) as bf16
#pragma unroll
    for (int i = 0; i < 4; ++i) {
      int g  = tid + i * 256;   // 32 keys x 32 d-quads
      int kk = g >> 5;
      int d0 = (g & 31) * 4;
      const float* kp = qkv + (size_t)(b * SQ + k0 + kk) * E3 + EE + h * DD + d0;
      v4f kv = *(const v4f*)kp;
      Kt[(d0 + 0) * 32 + kk] = (__bf16)kv.x;
      Kt[(d0 + 1) * 32 + kk] = (__bf16)kv.y;
      Kt[(d0 + 2) * 32 + kk] = (__bf16)kv.z;
      Kt[(d0 + 3) * 32 + kk] = (__bf16)kv.w;
      const float* vp = qkv + (size_t)(b * SQ + k0 + kk) * E3 + 2 * EE + h * DD + d0;
      v4f vv = *(const v4f*)vp;
      __bf16* vd = &Vs[kk * DD + d0];
      vd[0] = (__bf16)vv.x; vd[1] = (__bf16)vv.y;
      vd[2] = (__bf16)vv.z; vd[3] = (__bf16)vv.w;
    }
    __syncthreads();

    // ---- scores: S(16x32) = Q(16x128) * K^T, 2 n-tiles x 4 K-slices
    v8f sa[2]; sa[0] = zero; sa[1] = zero;
#pragma unroll
    for (int nt = 0; nt < 2; ++nt)
#pragma unroll
      for (int j = 0; j < 4; ++j) {
        const __bf16* bp = &Kt[(j * 32) * 32 + nt * 16 + m];
        v16bf kf;
#pragma unroll
        for (int s = 0; s < 16; ++s) kf[s] = bp[(kb + (s & 7) + ((s >> 3) << 4)) * 32];
        sa[nt] = wmma_bf16(qfr[j], kf, sa[nt]);
      }

    // ---- online softmax (fp32) on accumulator layout; P -> bf16 via LDS
#pragma unroll
    for (int r = 0; r < 8; ++r) {
      int rowg = q0 + r + 8 * half;
      int key0 = k0 + (lane & 15);
      float s0 = sa[0][r] * scale; if (key0      > rowg) s0 = -INFINITY;
      float s1 = sa[1][r] * scale; if (key0 + 16 > rowg) s1 = -INFINITY;
      float mx = fmaxf(s0, s1);
      mx = fmaxf(mx, __shfl_xor(mx, 1, 32));
      mx = fmaxf(mx, __shfl_xor(mx, 2, 32));
      mx = fmaxf(mx, __shfl_xor(mx, 4, 32));
      mx = fmaxf(mx, __shfl_xor(mx, 8, 32));
      float mnew = fmaxf(mrun[r], mx);
      float corr = __expf(mrun[r] - mnew);
      float p0 = __expf(s0 - mnew);
      float p1 = __expf(s1 - mnew);
      float rs = p0 + p1;
      rs += __shfl_xor(rs, 1, 32);
      rs += __shfl_xor(rs, 2, 32);
      rs += __shfl_xor(rs, 4, 32);
      rs += __shfl_xor(rs, 8, 32);
      lrun[r] = lrun[r] * corr + rs;
      mrun[r] = mnew;
#pragma unroll
      for (int dt = 0; dt < 8; ++dt) oacc[dt][r] *= corr;
      Ps[wave][(r + 8 * half) * 32 + (lane & 15)]      = (__bf16)p0;
      Ps[wave][(r + 8 * half) * 32 + 16 + (lane & 15)] = (__bf16)p1;
    }
    // Ps is private to this wave; wave-level ordering is enough (DS in-order).
    wave_lds_fence();

    // ---- P fragment (A layout) and P x V accumulation over D tiles
    v16bf pfr = cat16(*(const v8bf*)&Ps[wave][m * 32 + kb],
                      *(const v8bf*)&Ps[wave][m * 32 + kb + 16]);
#pragma unroll
    for (int dt = 0; dt < 8; ++dt) {
      const __bf16* vp = &Vs[dt * 16 + m];
      v16bf vf;
#pragma unroll
      for (int s = 0; s < 16; ++s) vf[s] = vp[(kb + (s & 7) + ((s >> 3) << 4)) * DD];
      oacc[dt] = wmma_bf16(pfr, vf, oacc[dt]);
    }
    __syncthreads();  // Kt/Vs consumed by all waves before next staging
  }

  // ---- normalize and write ctx (f32, [total, E] with head offset)
#pragma unroll
  for (int r = 0; r < 8; ++r) {
    int rowg = q0 + r + 8 * half;
    float inv = 1.0f / lrun[r];
#pragma unroll
    for (int dt = 0; dt < 8; ++dt) {
      int d = dt * 16 + (lane & 15);
      ctx[(size_t)(b * SQ + rowg) * EE + h * DD + d] = oacc[dt][r] * inv;
    }
  }
}

// ---------------------------------------------------------------------------
// Launch: QKV GEMM -> flash attention -> output projection.
// ws layout: qkv  [4096 x 6144] f32  (96 MiB)
//            ctx  [4096 x 2048] f32  (32 MiB)
// ---------------------------------------------------------------------------
extern "C" void kernel_launch(void* const* d_in, const int* in_sizes, int n_in,
                              void* d_out, int out_size, void* d_ws, size_t ws_size,
                              hipStream_t stream) {
  (void)in_sizes; (void)n_in; (void)out_size; (void)ws_size;
  const float* hidden = (const float*)d_in[0];
  const float* wqkv   = (const float*)d_in[1];
  const float* bqkv   = (const float*)d_in[2];
  const float* wproj  = (const float*)d_in[3];
  const float* bproj  = (const float*)d_in[4];
  // inputs 5..8 (key_cache, value_cache, slots, seq_len) are an identity
  // round-trip in the reference (slots == arange) -> not needed for `out`.

  float* qkv = (float*)d_ws;
  float* ctx = qkv + (size_t)4096 * 6144;

  dim3 blk(256);
  // qkv = hidden @ c_attn_w + c_attn_b   : M=4096 N=6144 K=2048
  gemm_bias_bf16wmma<<<dim3(6144 / BN, 4096 / BM), blk, 0, stream>>>(
      hidden, wqkv, bqkv, qkv, 4096, 6144, 2048);
  // ctx = softmax(QK^T * scale, causal) @ V
  flash_attn_bf16wmma<<<dim3(SQ / 128, 4 * 16), blk, 0, stream>>>(qkv, ctx);
  // out = ctx @ c_proj_w + c_proj_b      : M=4096 N=2048 K=2048
  gemm_bias_bf16wmma<<<dim3(2048 / BN, 4096 / BM), blk, 0, stream>>>(
      ctx, wproj, bproj, (float*)d_out, 4096, 2048, 2048);
}
